// KANLinear_644245094588
// MI455X (gfx1250) — matrix operand order repaired and checked by reading, not compiled, not run on hardware
//
#include <hip/hip_runtime.h>
#include <hip/hip_bf16.h>
#include <hip/hip_fp16.h>
#include <stdint.h>

// ---------------- problem constants ----------------
#define N_ROWS 8192
#define IN_F   1024
#define OUT_F  1024
#define GPTS   5
#define KDIM   (IN_F * GPTS)          // 5120 (K of the expanded GEMM)

// GEMM staging parameters
#define KB     64                     // K-halves per LDS stage (2 WMMA k-steps)
#define PAD    8                      // +8 halves (16B) row pad: bank step 36 mod 64 -> conflict-free
#define LROW   (KB + PAD)             // 72 halves per LDS row

typedef _Float16 half8 __attribute__((ext_vector_type(8)));
typedef _Float16 v16h  __attribute__((ext_vector_type(16)));
typedef float    v8f   __attribute__((ext_vector_type(8)));
typedef int      kan_i32x4 __attribute__((__vector_size__(16)));

#define KAN_AS1 __attribute__((address_space(1)))
#define KAN_AS3 __attribute__((address_space(3)))

// ---------------- async global->LDS plumbing (gfx1250 ASYNCcnt path) -------
#if defined(__has_builtin)
#  if __has_builtin(__builtin_amdgcn_global_load_async_to_lds_b128) && \
      __has_builtin(__builtin_amdgcn_s_wait_asynccnt)
#    define KAN_ASYNC_LDS 1
#  endif
#endif
#ifndef KAN_ASYNC_LDS
#  define KAN_ASYNC_LDS 0
#endif

__device__ __forceinline__ void kan_cp16_to_lds(_Float16* lds, const _Float16* g)
{
#if KAN_ASYNC_LDS
    // GLOBAL_LOAD_ASYNC_TO_LDS_B128: per-lane 16B global -> LDS, tracked by ASYNCcnt.
    // Builtin params are typed i32x4* (b128 payload). Generic->AS casts via integer
    // (low 32 bits of a generic LDS pointer = LDS offset per aperture rules).
    __builtin_amdgcn_global_load_async_to_lds_b128(
        (KAN_AS1 kan_i32x4*)(uintptr_t)g,
        (KAN_AS3 kan_i32x4*)(uint32_t)(uintptr_t)lds,
        0, 0);
#else
    *(half8*)lds = *(const half8*)g;   // sync fallback: global_load_b128 + ds_store_b128
#endif
}

#if KAN_ASYNC_LDS
#  define KAN_WAIT_ASYNC(n) __builtin_amdgcn_s_wait_asynccnt(n)
#else
#  define KAN_WAIT_ASYNC(n) do {} while (0)
#endif

// ---------------------------------------------------------------------------
// Kernel 1: build f16 coeff matrix A[n][k], k = g*IN_F + i.
// coeff[n, i, g] = (1-t) at g==idx, t at g==idx+1, else 0.
// grid = linspace(-1,1,5): step 0.5, so s = (tanh(x)+1)*2 is the cell coord.
// ---------------------------------------------------------------------------
__global__ void kan_pack_coeff(const float* __restrict__ x,
                               _Float16* __restrict__ A)
{
    const int tid = blockIdx.x * blockDim.x + threadIdx.x;   // over N_ROWS*IN_F
    const int n = tid >> 10;              // IN_F = 1024
    const int i = tid & (IN_F - 1);

    const float xc = tanhf(x[tid]);       // already in [-1,1] -> clip is no-op
    const float s  = (xc + 1.0f) * 2.0f;  // position in units of grid step 0.5
    int idx = (int)s;                     // s in [0,4]
    idx = idx > 3 ? 3 : (idx < 0 ? 0 : idx);
    const float t  = s - (float)idx;      // (x_c - g_left)/0.5
    const float cl = 1.0f - t;

    _Float16* row = A + (size_t)n * KDIM + i;
#pragma unroll
    for (int g = 0; g < GPTS; ++g) {
        float v = (g == idx) ? cl : ((g == idx + 1) ? t : 0.0f);
        row[(size_t)g * IN_F] = (_Float16)v;
    }
}

// ---------------------------------------------------------------------------
// Kernel 2: Bp[o][k] = (f16) W[o][i][g] with k = g*IN_F + i.
// Bp is 10.5 MB -> resident in the 192 MB L2 for the whole GEMM.
// ---------------------------------------------------------------------------
__global__ void kan_pack_w(const float* __restrict__ W,
                           _Float16* __restrict__ Bp)
{
    const int tid = blockIdx.x * blockDim.x + threadIdx.x;   // over OUT_F*KDIM
    const int o = tid / KDIM;
    const int r = tid - o * KDIM;
    const int g = r >> 10;                // IN_F = 1024
    const int i = r & (IN_F - 1);
    Bp[tid] = (_Float16)W[(size_t)o * (IN_F * GPTS) + (size_t)i * GPTS + g];
}

// ---------------------------------------------------------------------------
// Kernel 3: C[N][O] = A[N][K] * Bp[O][K]^T  (f16 inputs, f32 accumulate)
//
// Block tile 128(M) x 128(N), 256 threads = 8 wave32 waves laid out 4(M)x2(N),
// each wave owns 32x64 = 2x4 v_wmma_f32_16x16x32_f16 accumulators.
//
// Double-buffered LDS pipeline: stage = 128x64 A-panel + 128x64 B-panel,
// copied with GLOBAL_LOAD_ASYNC_TO_LDS_B128 (8 ops/thread/stage), prefetching
// stage s+1 while computing stage s; sync = s_wait_asynccnt 8 + barrier.
//
// Fragment layouts per CDNA5 ISA (CDNA5 ISA):
//  A 16x32 f16 : lane L holds M = L%16; halves e: K = 8*(L/16) + (e<8 ? e : e+8)
//  B 32x16 f16 : lane L holds N = L%16; halves e: K = 8*(L/16) + (e<8 ? e : e+8)
//  C 16x16 f32 : vgpr r: M = r + 8*(L/16), N = L%16
// LDS rows padded to 72 halves (144B): fragment reads cover all 64 banks.
// ---------------------------------------------------------------------------
__global__ __launch_bounds__(256)
void kan_wmma_gemm(const _Float16* __restrict__ A,
                   const _Float16* __restrict__ Bp,
                   float* __restrict__ C)
{
    __shared__ _Float16 sA[2][128][LROW];   // 36 KB
    __shared__ _Float16 sB[2][128][LROW];   // 36 KB

    const int tid  = threadIdx.x;
    const int lane = tid & 31;
    const int wave = tid >> 5;
    const int wm   = wave & 3;            // 0..3 : M sub-tile
    const int wn   = wave >> 2;           // 0..1 : N sub-tile
    const int lrow = lane & 15;
    const int lhi  = lane >> 4;           // 0/1 : which half-wave

    const int bm0 = blockIdx.y * 128;
    const int bn0 = blockIdx.x * 128;

    // Stage copy: 1024 16-byte chunks per panel (128 rows x 8 chunks),
    // 4 chunks of A + 4 chunks of B per thread.
    auto stage_copy = [&](int s, int kbase) {
#pragma unroll
        for (int j = 0; j < 4; ++j) {
            const int c   = tid + j * 256;
            const int row = c >> 3;
            const int kc  = (c & 7) * 8;
            kan_cp16_to_lds(&sA[s][row][kc],
                            A + (size_t)(bm0 + row) * KDIM + kbase + kc);
        }
#pragma unroll
        for (int j = 0; j < 4; ++j) {
            const int c   = tid + j * 256;
            const int row = c >> 3;
            const int kc  = (c & 7) * 8;
            kan_cp16_to_lds(&sB[s][row][kc],
                            Bp + (size_t)(bn0 + row) * KDIM + kbase + kc);
        }
    };

    v8f acc[2][4];
#pragma unroll
    for (int mf = 0; mf < 2; ++mf)
#pragma unroll
        for (int nf = 0; nf < 4; ++nf)
            acc[mf][nf] = (v8f)(0.0f);

    const int NT = KDIM / KB;             // 80 stages
    stage_copy(0, 0);

    for (int it = 0; it < NT; ++it) {
        const int s = it & 1;
        if (it + 1 < NT) {
            stage_copy(s ^ 1, (it + 1) * KB);  // prefetch next stage (other buffer)
            KAN_WAIT_ASYNC(8);                 // this stage's 8 copies complete
        } else {
            KAN_WAIT_ASYNC(0);
        }
        __syncthreads();                       // all threads' copies visible

#pragma unroll
        for (int kk = 0; kk < 2; ++kk) {       // two WMMA k-steps per stage
            const int k0 = kk * 32;
            v16h afrag[2];
#pragma unroll
            for (int mf = 0; mf < 2; ++mf) {
                const _Float16* p = &sA[s][wm * 32 + mf * 16 + lrow][k0 + 8 * lhi];
                half8 lo = *(const half8*)p;          // K: e=0..7 block
                half8 hi = *(const half8*)(p + 16);   // K: e=8..15 block
                afrag[mf] = __builtin_shufflevector(lo, hi,
                    0, 1, 2, 3, 4, 5, 6, 7, 8, 9, 10, 11, 12, 13, 14, 15);
            }
            v16h bfrag[4];
#pragma unroll
            for (int nf = 0; nf < 4; ++nf) {
                const _Float16* p = &sB[s][wn * 64 + nf * 16 + lrow][k0 + 8 * lhi];
                half8 lo = *(const half8*)p;
                half8 hi = *(const half8*)(p + 16);
                bfrag[nf] = __builtin_shufflevector(lo, hi,
                    0, 1, 2, 3, 4, 5, 6, 7, 8, 9, 10, 11, 12, 13, 14, 15);
            }
#pragma unroll
            for (int mf = 0; mf < 2; ++mf)
#pragma unroll
                for (int nf = 0; nf < 4; ++nf)
                    acc[mf][nf] = __builtin_amdgcn_wmma_f32_16x16x32_f16(
                        false, afrag[mf], false, bfrag[nf],
                        (short)0, acc[mf][nf], false, false);
        }
        __syncthreads();                       // stage fully consumed before reuse
    }

    // Epilogue: C layout vgpr r -> M = r + 8*lhi, N = lrow
#pragma unroll
    for (int mf = 0; mf < 2; ++mf) {
#pragma unroll
        for (int nf = 0; nf < 4; ++nf) {
            const int col = bn0 + wn * 64 + nf * 16 + lrow;
#pragma unroll
            for (int r = 0; r < 8; ++r) {
                const int row = bm0 + wm * 32 + mf * 16 + r + 8 * lhi;
                C[(size_t)row * OUT_F + col] = acc[mf][nf][r];
            }
        }
    }
}

// ---------------------------------------------------------------------------
// Host-side launcher. Workspace layout:
//   A  : N_ROWS * KDIM f16  = 83.9 MB
//   Bp : OUT_F  * KDIM f16  = 10.5 MB   (total ~94.4 MB, assumed <= ws_size)
// ---------------------------------------------------------------------------
extern "C" void kernel_launch(void* const* d_in, const int* in_sizes, int n_in,
                              void* d_out, int out_size, void* d_ws, size_t ws_size,
                              hipStream_t stream)
{
    const float* x = (const float*)d_in[0];                 // [8192, 1024] f32
    const float* W = (const float*)d_in[1];                 // [1024, 1024, 5] f32
    float* out = (float*)d_out;                             // [8192, 1024] f32

    _Float16* A  = (_Float16*)d_ws;
    _Float16* Bp = (_Float16*)((char*)d_ws + (size_t)N_ROWS * KDIM * sizeof(_Float16));

    {   // coeff build: one thread per (n, i)
        const int total = N_ROWS * IN_F;
        kan_pack_coeff<<<total / 256, 256, 0, stream>>>(x, A);
    }
    {   // weight convert/permute: one thread per (o, k)
        const int total = OUT_F * KDIM;
        kan_pack_w<<<total / 256, 256, 0, stream>>>(W, Bp);
    }
    {   // WMMA GEMM: grid.x over OUT_F tiles, grid.y over N tiles
        dim3 grid(OUT_F / 128, N_ROWS / 128);
        kan_wmma_gemm<<<grid, 256, 0, stream>>>(A, Bp, out);
    }
}
